// SparseCostRegNet_45870250721614
// MI455X (gfx1250) — matrix operand order repaired and checked by reading, not compiled, not run on hardware
//
#include <hip/hip_runtime.h>

typedef __attribute__((ext_vector_type(16))) __bf16 v16bf;
typedef __attribute__((ext_vector_type(8)))  __bf16 v8bf;
typedef __attribute__((ext_vector_type(8)))  float  v8f;

// ---------------------------------------------------------------------------
// Utility kernels
// ---------------------------------------------------------------------------
__global__ void zero_f32(float* p, int n) {
  int i = blockIdx.x * blockDim.x + threadIdx.x;
  if (i < n) p[i] = 0.f;
}

__global__ void f32_to_bf16(const float* __restrict__ src,
                            __bf16* __restrict__ dst, int n) {
  int i = blockIdx.x * blockDim.x + threadIdx.x;
  if (i < n) dst[i] = (__bf16)src[i];
}

// Swizzle f32 weights [27, Cin, Cout] into the contiguous-k WMMA B-register order
// (bf16).  Block layout: dst[((t*KC + j)*32 + lane)*16 + idx], where the 16
// bf16 for lane L are K = 32*j + (L>>4)*16 + idx, column n = (L&15) + 16*t.
__global__ void prep_w(const float* __restrict__ w, __bf16* __restrict__ dst,
                       int cin, int lc, int cout, int T, int kc) {
  int total = T * kc * 512;
  int e = blockIdx.x * blockDim.x + threadIdx.x;
  if (e >= total) return;
  int idx  = e & 15;
  int lane = (e >> 4) & 31;
  int rest = e >> 9;
  int j = rest % kc;
  int t = rest / kc;
  int n  = (lane & 15) + (t << 4);
  int hi = lane >> 4;
  int kflat = j * 32 + hi * 16 + idx;
  float v = 0.f;
  if (kflat < 27 * cin && n < cout) {
    int k = kflat >> lc;
    int c = kflat & (cin - 1);
    v = w[((size_t)k * cin + c) * cout + n];
  }
  dst[e] = (__bf16)v;
}

// ---------------------------------------------------------------------------
// Sparse conv as gather + bf16 WMMA GEMM, with per-channel sum/sumsq side
// accumulation for training-mode BatchNorm.
//   fh   : input features [Nin, CIN] bf16 (pre-converted)
//   nbr  : [Nout, 27] neighbor row indices into fh (-1 = absent)
//   Bw   : pre-swizzled bf16 weights (see prep_w)
//   y    : [Nout, COUT] raw conv output (f32)
//   stats: [2*COUT] f32, {sum, sumsq}, pre-zeroed
// ---------------------------------------------------------------------------
template <int CIN, int COUT>
__global__ void __launch_bounds__(256)
sconv_wmma(const __bf16* __restrict__ fh, const int* __restrict__ nbr,
           const __bf16* __restrict__ Bw, float* __restrict__ y,
           float* __restrict__ stats, int Nout) {
  constexpr int KTOT  = 27 * CIN;
  constexpr int KC    = (KTOT + 31) / 32;
  constexpr int T     = (COUT + 15) / 16;
  constexpr int LC    = (CIN == 8) ? 3 : (CIN == 16) ? 4 : (CIN == 32) ? 5 : 6;
  constexpr int CPERK = (CIN < 16) ? CIN : 16;  // channels per 16-col segment
  constexpr int RUNK  = 16 / CPERK;             // segments per lane per chunk

  __shared__ alignas(16) __bf16 Atile[8][16][32];  // per-wave A staging
  __shared__ float ssum[64], ssq[64];

  const int tid  = threadIdx.x;
  const int wid  = tid >> 5;
  const int lane = tid & 31;
  const int hi   = lane >> 4;
  const int ln   = lane & 15;

  if (tid < 64) { ssum[tid] = 0.f; ssq[tid] = 0.f; }
  __syncthreads();

  const int rowbase = blockIdx.x * 128 + wid * 16;

  // Gather lane mapping: lane -> (row 0..15, 16-col half)
  const int gr    = lane >> 1;
  const int gcb   = (lane & 1) << 4;
  const int grow  = rowbase + gr;
  const bool gval = (grow < Nout);
  const int* nrow = nbr + (long long)grow * 27;
  if (gval) __builtin_prefetch(nrow, 0, 3);   // global_prefetch_b8

  const v8f zero8 = {0.f, 0.f, 0.f, 0.f, 0.f, 0.f, 0.f, 0.f};
  v8f acc[T];
#pragma unroll
  for (int t = 0; t < T; ++t) acc[t] = zero8;

  for (int j = 0; j < KC; ++j) {
    const int kbase = j * 32;
    // ---- vectorized gather: bf16 rows -> LDS via b128 ----
#pragma unroll
    for (int s = 0; s < RUNK; ++s) {
      const int cbase = gcb + s * CPERK;       // column within 32-wide chunk
      const int base  = kbase + cbase;         // flattened (k,cin) index
      const int k     = base >> LC;            // stencil offset (seg in one k)
      uint4 d0 = {0u, 0u, 0u, 0u};
      uint4 d1 = {0u, 0u, 0u, 0u};
      if (gval && k < 27) {
        const int src = nrow[k];
        if (src >= 0) {
          const uint4* sp = (const uint4*)(fh + (long long)src * CIN +
                                           (base & (CIN - 1)));
          d0 = sp[0];
          if (CPERK == 16) d1 = sp[1];
        }
      }
      uint4* dp = (uint4*)(&Atile[wid][gr][cbase]);
      dp[0] = d0;
      if (CPERK == 16) dp[1] = d1;
    }
    __builtin_amdgcn_wave_barrier();   // wave-local staging; DScnt orders ds ops

    // ---- build A fragment (contiguous k = 16*hi + s, as in prep_w) ----
    const v8bf* rowp = (const v8bf*)(&Atile[wid][ln][0]);
    v8bf a0 = rowp[2 * hi];       // K = hi*16 .. hi*16+7
    v8bf a1 = rowp[2 * hi + 1];   // K = hi*16+8 .. hi*16+15
    v16bf av = __builtin_shufflevector(a0, a1, 0, 1, 2, 3, 4, 5, 6, 7,
                                       8, 9, 10, 11, 12, 13, 14, 15);

    // ---- WMMA over Cout tiles ----
#pragma unroll
    for (int t = 0; t < T; ++t) {
      const v8bf* bp =
          (const v8bf*)(Bw + (((long long)t * KC + j) * 32 + lane) * 16);
      v8bf b0 = bp[0], b1 = bp[1];
      v16bf bv = __builtin_shufflevector(b0, b1, 0, 1, 2, 3, 4, 5, 6, 7,
                                         8, 9, 10, 11, 12, 13, 14, 15);
      acc[t] = __builtin_amdgcn_wmma_f32_16x16x32_bf16(
          false, av, false, bv, (short)0, acc[t], false, false);
    }
    __builtin_amdgcn_wave_barrier();
  }

  // ---- store y and accumulate BN statistics ----
#pragma unroll
  for (int t = 0; t < T; ++t) {
    const int ch = t * 16 + ln;
    float ls = 0.f, ls2 = 0.f;
#pragma unroll
    for (int r = 0; r < 8; ++r) {
      const int m = r + hi * 8;
      const int row = rowbase + m;
      const float v = acc[t][r];
      if (row < Nout && ch < COUT) {
        y[(long long)row * COUT + ch] = v;
        ls += v;
        ls2 += v * v;
      }
    }
    if (ch < COUT) {
      atomicAdd(&ssum[ch], ls);    // ds_add_f32
      atomicAdd(&ssq[ch], ls2);
    }
  }
  __syncthreads();
  if (tid < COUT) {
    atomicAdd(&stats[tid], ssum[tid]);          // global_atomic_add_f32
    atomicAdd(&stats[COUT + tid], ssq[tid]);
  }
}

// Fold BN (training-mode) + gamma/beta into per-channel affine: y*sc + sh
__global__ void bnparam(const float* __restrict__ stats,
                        const float* __restrict__ g, const float* __restrict__ b,
                        float* __restrict__ prm, int cout, float invN) {
  int c = threadIdx.x;
  if (c >= cout) return;
  float mu  = stats[c] * invN;
  float var = stats[cout + c] * invN - mu * mu;
  float sc  = g[c] * rsqrtf(var + 1e-5f);
  prm[c] = sc;
  prm[cout + c] = b[c] - mu * sc;
}

// out = (res? res : 0) + relu(y*sc + sh); dual-write f32 + bf16
__global__ void bn_relu_res(const float* __restrict__ yv,
                            const float* __restrict__ prm,
                            const float* __restrict__ res,
                            float* __restrict__ out,
                            __bf16* __restrict__ outh, int total, int cmask,
                            int cout) {
  int i = blockIdx.x * blockDim.x + threadIdx.x;
  if (i >= total) return;
  int c = i & cmask;
  float v = yv[i] * prm[c] + prm[cout + c];
  v = v > 0.f ? v : 0.f;
  if (res) v += res[i];
  out[i] = v;
  outh[i] = (__bf16)v;
}

// out0 = x @ lin_w.T  ([8,8]);  out1 = x
__global__ void final_linear(const float* __restrict__ x,
                             const float* __restrict__ lw,
                             float* __restrict__ out, int n0) {
  int i = blockIdx.x * blockDim.x + threadIdx.x;
  if (i >= n0) return;
  float xv[8];
#pragma unroll
  for (int k = 0; k < 8; ++k) xv[k] = x[(size_t)i * 8 + k];
  float* out1 = out + (size_t)n0 * 8;
#pragma unroll
  for (int j = 0; j < 8; ++j) {
    float s = 0.f;
#pragma unroll
    for (int k = 0; k < 8; ++k) s += xv[k] * lw[j * 8 + k];
    out[(size_t)i * 8 + j]  = s;
    out1[(size_t)i * 8 + j] = xv[j];
  }
}

// ---------------------------------------------------------------------------
// Host orchestration
// ---------------------------------------------------------------------------
template <int CIN, int COUT>
static inline void run_layer(const __bf16* fh, const int* nbr,
                             const __bf16* Bw, float* ybuf, float* stats,
                             float* prm, const float* g, const float* b,
                             const float* res, float* aout, __bf16* aouth,
                             int Nout, hipStream_t s) {
  sconv_wmma<CIN, COUT><<<(Nout + 127) / 128, 256, 0, s>>>(fh, nbr, Bw, ybuf,
                                                           stats, Nout);
  bnparam<<<1, 64, 0, s>>>(stats, g, b, prm, COUT, 1.0f / (float)Nout);
  int total = Nout * COUT;
  bn_relu_res<<<(total + 255) / 256, 256, 0, s>>>(ybuf, prm, res, aout, aouth,
                                                  total, COUT - 1, COUT);
}

extern "C" void kernel_launch(void* const* d_in, const int* in_sizes, int n_in,
                              void* d_out, int out_size, void* d_ws,
                              size_t ws_size, hipStream_t stream) {
  (void)n_in; (void)out_size; (void)ws_size;

  const float* feats = (const float*)d_in[0];
  const int* nbrs[10];
  for (int l = 0; l < 10; ++l) nbrs[l] = (const int*)d_in[1 + l];
  const float *W[10], *G[10], *B[10];
  for (int i = 0; i < 10; ++i) {
    W[i] = (const float*)d_in[11 + 3 * i];
    G[i] = (const float*)d_in[12 + 3 * i];
    B[i] = (const float*)d_in[13 + 3 * i];
  }
  const float* lin_w = (const float*)d_in[41];

  const int N0 = in_sizes[0] / 16;
  const int N1 = in_sizes[2] / 27;
  const int N2 = in_sizes[4] / 27;
  const int N3 = in_sizes[6] / 27;
  const int Nout[10] = {N0, N1, N1, N2, N2, N3, N3, N2, N1, N0};
  static const int CIa[10] = {16, 8, 16, 16, 32, 32, 64, 64, 32, 16};
  static const int COa[10] = {8, 16, 16, 32, 32, 64, 64, 32, 16, 8};
  static const int LCa[10] = {4, 3, 4, 4, 5, 5, 6, 6, 5, 4};

  // ---- bump-allocate workspace ----
  char* ws = (char*)d_ws;
  size_t off = 0;
  auto alloc = [&](size_t bytes) -> char* {
    char* p = ws + off;
    off = (off + bytes + 255) & ~(size_t)255;
    return p;
  };

  int statsTot = 0;
  for (int l = 0; l < 10; ++l) statsTot += 2 * COa[l];
  float* statsAll = (float*)alloc((size_t)statsTot * 4);
  float* prmAll   = (float*)alloc((size_t)statsTot * 4);
  float* stats[10]; float* prm[10];
  { int o = 0;
    for (int l = 0; l < 10; ++l) { stats[l] = statsAll + o; prm[l] = prmAll + o; o += 2 * COa[l]; } }

  __bf16* Bw[10];
  int Ta[10], KCa[10];
  for (int l = 0; l < 10; ++l) {
    KCa[l] = (27 * CIa[l] + 31) / 32;
    Ta[l]  = (COa[l] + 15) / 16;
    Bw[l]  = (__bf16*)alloc((size_t)Ta[l] * KCa[l] * 512 * 2);
  }

  float* act[10]; __bf16* acth[10];
  for (int l = 0; l < 10; ++l) {
    size_t n = (size_t)Nout[l] * COa[l];
    act[l]  = (float*)alloc(n * 4);
    acth[l] = (__bf16*)alloc(n * 2);
  }
  __bf16* featsh = (__bf16*)alloc((size_t)N0 * 16 * 2);
  size_t ymax = 0;
  for (int l = 0; l < 10; ++l) {
    size_t n = (size_t)Nout[l] * COa[l];
    if (n > ymax) ymax = n;
  }
  float* ybuf = (float*)alloc(ymax * 4);

  // ---- per-call init (graph-replay safe) ----
  zero_f32<<<(statsTot + 255) / 256, 256, 0, stream>>>(statsAll, statsTot);
  f32_to_bf16<<<(N0 * 16 + 255) / 256, 256, 0, stream>>>(feats, featsh,
                                                         N0 * 16);
  for (int l = 0; l < 10; ++l) {
    int total = Ta[l] * KCa[l] * 512;
    prep_w<<<(total + 255) / 256, 256, 0, stream>>>(W[l], Bw[l], CIa[l], LCa[l],
                                                    COa[l], Ta[l], KCa[l]);
  }

  // ---- U-Net ----
  run_layer<16, 8 >(featsh,  nbrs[0], Bw[0], ybuf, stats[0], prm[0], G[0], B[0], nullptr, act[0], acth[0], Nout[0], stream);
  run_layer<8 , 16>(acth[0], nbrs[1], Bw[1], ybuf, stats[1], prm[1], G[1], B[1], nullptr, act[1], acth[1], Nout[1], stream);
  run_layer<16, 16>(acth[1], nbrs[2], Bw[2], ybuf, stats[2], prm[2], G[2], B[2], nullptr, act[2], acth[2], Nout[2], stream);
  run_layer<16, 32>(acth[2], nbrs[3], Bw[3], ybuf, stats[3], prm[3], G[3], B[3], nullptr, act[3], acth[3], Nout[3], stream);
  run_layer<32, 32>(acth[3], nbrs[4], Bw[4], ybuf, stats[4], prm[4], G[4], B[4], nullptr, act[4], acth[4], Nout[4], stream);
  run_layer<32, 64>(acth[4], nbrs[5], Bw[5], ybuf, stats[5], prm[5], G[5], B[5], nullptr, act[5], acth[5], Nout[5], stream);
  run_layer<64, 64>(acth[5], nbrs[6], Bw[6], ybuf, stats[6], prm[6], G[6], B[6], nullptr, act[6], acth[6], Nout[6], stream);
  run_layer<64, 32>(acth[6], nbrs[7], Bw[7], ybuf, stats[7], prm[7], G[7], B[7], act[4], act[7], acth[7], Nout[7], stream);
  run_layer<32, 16>(acth[7], nbrs[8], Bw[8], ybuf, stats[8], prm[8], G[8], B[8], act[2], act[8], acth[8], Nout[8], stream);
  run_layer<16, 8 >(acth[8], nbrs[9], Bw[9], ybuf, stats[9], prm[9], G[9], B[9], act[0], act[9], acth[9], Nout[9], stream);

  final_linear<<<(N0 + 255) / 256, 256, 0, stream>>>(act[9], lin_w,
                                                     (float*)d_out, N0);
}
